// semantic_50216757625345
// MI455X (gfx1250) — compile-verified, matrix operand order checked by reading, not run on hardware
//
#include <hip/hip_runtime.h>
#include <hip/hip_bf16.h>
#include <math.h>

// Problem constants
#define BB   8
#define CC   2048
#define HH   14
#define WW   14
#define SS   (HH*WW)          // 196
#define SB   (SS*BB)          // 1568
#define SBP  1600             // padded M for GEMM (multiple of 64)
#define NC   80
#define WD   300
#define DD   1024

typedef __attribute__((ext_vector_type(16))) __bf16 v16bf;
typedef __attribute__((ext_vector_type(8)))  __bf16 v8bf;
typedef __attribute__((ext_vector_type(8)))  float  v8f;

// Fast hardware tanh: CDNA5 has V_TANH_F32 (trans op). Prefer the clang
// builtin if this toolchain declares it; otherwise emit the instruction
// directly with inline asm.
__device__ __forceinline__ float fast_tanh(float x) {
#if defined(__has_builtin) && __has_builtin(__builtin_amdgcn_tanhf)
    return __builtin_amdgcn_tanhf(x);
#elif defined(__has_builtin) && __has_builtin(__builtin_amdgcn_tanh_f32)
    return __builtin_amdgcn_tanh_f32(x);
#else
    float r;
    asm volatile("v_tanh_f32 %0, %1" : "=v"(r) : "v"(x));
    return r;
#endif
}

// ---------------------------------------------------------------------------
// Prep: v[d] = sum_e Wa[e] * W3[e,d]
__global__ void k_prep_v(const float* __restrict__ W3, const float* __restrict__ Wa,
                         float* __restrict__ v) {
    int d = blockIdx.x * blockDim.x + threadIdx.x;
    if (d >= DD) return;
    float acc = 0.f;
    for (int e = 0; e < DD; ++e)
        acc = fmaf(Wa[e], W3[e * DD + d], acc);   // coalesced across d
    v[d] = acc;
}

// c0 = b3 . Wa + ba  (one wave)
__global__ void k_prep_c0(const float* __restrict__ b3, const float* __restrict__ Wa,
                          const float* __restrict__ ba, float* __restrict__ c0) {
    int l = threadIdx.x;
    float acc = 0.f;
    for (int d = l; d < DD; d += 32) acc = fmaf(b3[d], Wa[d], acc);
    for (int off = 16; off > 0; off >>= 1) acc += __shfl_down(acc, off, 32);
    if (l == 0) c0[0] = acc + ba[0];
}

// f_wd[n,d] = sum_w word[n,w] * W2[d,w]
__global__ void k_prep_fwd(const float* __restrict__ word, const float* __restrict__ W2,
                           float* __restrict__ f_wd) {
    int idx = blockIdx.x * blockDim.x + threadIdx.x;      // n*DD + d
    if (idx >= NC * DD) return;
    int n = idx / DD, d = idx % DD;
    const float* wr = word + n * WD;
    const float* w2 = W2 + d * WD;
    float acc = 0.f;
    for (int w = 0; w < WD; ++w) acc = fmaf(wr[w], w2[w], acc);
    f_wd[idx] = acc;
}

// ---------------------------------------------------------------------------
// Pack W1 (f32 row-major [D][C]) -> bf16
__global__ void k_pack_w1(const float* __restrict__ W1, __bf16* __restrict__ W1b) {
    int i = blockIdx.x * blockDim.x + threadIdx.x;
    if (i < DD * CC) W1b[i] = (__bf16)W1[i];
}

// Transpose image_features[b, c, s] -> img_t[sb = s*8+b, c] (bf16), zero-pad sb in [1568,1600)
__global__ void k_pack_img(const float* __restrict__ img, __bf16* __restrict__ img_t) {
    __shared__ float tile[32][33];
    const int s0 = blockIdx.x * 32;
    const int c0 = blockIdx.y * 32;
    const int b  = blockIdx.z;
    const int tx = threadIdx.x;       // 0..31 (s-local on read, c-local on write)
    const int ty = threadIdx.y;       // 0..7
#pragma unroll
    for (int i = 0; i < 4; ++i) {
        int cl = ty + i * 8;
        int s  = s0 + tx;
        float val = (s < SS) ? img[((size_t)(b * CC + c0 + cl)) * SS + s] : 0.f;
        tile[cl][tx] = val;           // coalesced read over s
    }
    __syncthreads();
#pragma unroll
    for (int j = 0; j < 4; ++j) {
        int sl = ty + j * 8;
        int s  = s0 + sl;
        if (s < 200) {                // sb = s*8+b < 1600
            int sb = s * BB + b;
            float val = (s < SS) ? tile[tx][sl] : 0.f;
            img_t[(size_t)sb * CC + (c0 + tx)] = (__bf16)val;  // coalesced over c
        }
    }
}

// ---------------------------------------------------------------------------
// WMMA bf16 GEMM: f_wh[m, n] = sum_k img_t[m, k] * W1[n, k]
// M=1600 (padded), N=1024, K=2048. Block: 8 waves (256 thr), block tile 64x128,
// wave tile 32x32 (2x2 WMMA accumulators).
__device__ __forceinline__ v16bf load_frag(const __bf16* p) {
    v8bf lo = *(const v8bf*)(p);        // K group +0..7
    v8bf hi = *(const v8bf*)(p + 16);   // K group +16..23
    v16bf r;
#pragma unroll
    for (int i = 0; i < 8; ++i) { r[i] = lo[i]; r[i + 8] = hi[i]; }
    return r;
}

__global__ void k_gemm_fwh(const __bf16* __restrict__ A,   // img_t [SBP][CC]
                           const __bf16* __restrict__ Bm,  // W1b   [DD][CC]
                           float* __restrict__ Cout) {     // f_wh  [SBP][DD]
    const int tid = threadIdx.x;
    const int wv  = tid >> 5;           // 0..7
    const int l   = tid & 31;
    const int lm  = l & 15;
    const int kh  = (l >= 16) ? 8 : 0;  // K-half select per ISA 16-bit A/B layout

    const int mbase = blockIdx.y * 64 + (wv >> 2) * 32;   // wave rows: 2
    const int nbase = blockIdx.x * 128 + (wv & 3) * 32;   // wave cols: 4

    v8f c00 = {}, c01 = {}, c10 = {}, c11 = {};

    const __bf16* a0p = A  + (size_t)(mbase + lm)      * CC + kh;
    const __bf16* a1p = A  + (size_t)(mbase + 16 + lm) * CC + kh;
    const __bf16* b0p = Bm + (size_t)(nbase + lm)      * CC + kh;
    const __bf16* b1p = Bm + (size_t)(nbase + 16 + lm) * CC + kh;

    for (int k0 = 0; k0 < CC; k0 += 32) {
        v16bf a0 = load_frag(a0p + k0);
        v16bf a1 = load_frag(a1p + k0);
        v16bf b0 = load_frag(b0p + k0);
        v16bf b1 = load_frag(b1p + k0);
        c00 = __builtin_amdgcn_wmma_f32_16x16x32_bf16(false, a0, false, b0, (short)0, c00, false, false);
        c01 = __builtin_amdgcn_wmma_f32_16x16x32_bf16(false, a0, false, b1, (short)0, c01, false, false);
        c10 = __builtin_amdgcn_wmma_f32_16x16x32_bf16(false, a1, false, b0, (short)0, c10, false, false);
        c11 = __builtin_amdgcn_wmma_f32_16x16x32_bf16(false, a1, false, b1, (short)0, c11, false, false);
    }

    // C/D layout: VGPR r holds row M = r + (lane>=16 ? 8 : 0), col N = lane%16
    const int rofs = (l >= 16) ? 8 : 0;
#pragma unroll
    for (int r = 0; r < 8; ++r) {
        int m0 = mbase + rofs + r;
        int m1 = mbase + 16 + rofs + r;
        Cout[(size_t)m0 * DD + nbase      + lm] = c00[r];
        Cout[(size_t)m0 * DD + nbase + 16 + lm] = c01[r];
        Cout[(size_t)m1 * DD + nbase      + lm] = c10[r];
        Cout[(size_t)m1 * DD + nbase + 16 + lm] = c11[r];
    }
}

// ---------------------------------------------------------------------------
// coef[sb, n] = c0 + sum_d tanh(f_wh[sb,d] * f_wd[n,d]) * v[d]
// One block per sb; f_wh row + v staged in LDS; 8 waves x 10 classes each.
// Dominant stage: 128M hardware v_tanh_f32 trans ops.
__global__ void k_coef(const float* __restrict__ f_wh, const float* __restrict__ f_wd,
                       const float* __restrict__ v, const float* __restrict__ c0,
                       float* __restrict__ coef) {
    __shared__ float fs[DD];
    __shared__ float vs[DD];
    const int sb  = blockIdx.x;
    const int tid = threadIdx.x;
    for (int i = tid; i < DD; i += 256) {
        fs[i] = f_wh[(size_t)sb * DD + i];
        vs[i] = v[i];
    }
    __syncthreads();
    const int wv = tid >> 5;
    const int l  = tid & 31;
    const float cc0 = c0[0];
#pragma unroll
    for (int t = 0; t < 10; ++t) {
        int n = wv * 10 + t;
        const float* fd = f_wd + (size_t)n * DD;
        float acc = 0.f;
        for (int d = l; d < DD; d += 32)
            acc = fmaf(fast_tanh(fs[d] * fd[d]), vs[d], acc);
        for (int off = 16; off > 0; off >>= 1) acc += __shfl_down(acc, off, 32);
        if (l == 0) coef[(size_t)sb * NC + n] = acc + cc0;
    }
}

// ---------------------------------------------------------------------------
// Softmax over s for each of 640 (b,n) columns; coef[s*640 + col], in place.
__global__ void k_softmax(float* __restrict__ coef) {
    int col = blockIdx.x * blockDim.x + threadIdx.x;
    if (col >= BB * NC) return;
    float mx = -1e30f;
    for (int s = 0; s < SS; ++s) mx = fmaxf(mx, coef[s * (BB * NC) + col]);
    float sum = 0.f;
    for (int s = 0; s < SS; ++s) {
        float e = expf(coef[s * (BB * NC) + col] - mx);
        coef[s * (BB * NC) + col] = e;
        sum += e;
    }
    float inv = 1.f / sum;
    for (int s = 0; s < SS; ++s) coef[s * (BB * NC) + col] *= inv;
}

// ---------------------------------------------------------------------------
// out[b, n, c] = sum_s image_features[b, c, s] * coef[s*640 + b*80 + n]
// img row is 196 f32 = 49 float4 (784B, 16B-aligned rows); stream as float4.
__global__ void k_out(const float* __restrict__ img, const float* __restrict__ coef,
                      float* __restrict__ out) {
    __shared__ float cs[SS];
    const int n = blockIdx.x;
    const int b = blockIdx.y;
    const int tid = threadIdx.x;
    if (tid < SS) cs[tid] = coef[tid * (BB * NC) + b * NC + n];
    __syncthreads();
#pragma unroll
    for (int i = 0; i < 8; ++i) {
        int c = tid * 8 + i;                       // 8 consecutive rows per thread
        const float4* row4 = (const float4*)(img + ((size_t)(b * CC + c)) * SS);
        float acc = 0.f;
#pragma unroll 7
        for (int s4 = 0; s4 < SS / 4; ++s4) {
            float4 r = row4[s4];
            const float* cp = &cs[s4 * 4];
            acc = fmaf(r.x, cp[0], acc);
            acc = fmaf(r.y, cp[1], acc);
            acc = fmaf(r.z, cp[2], acc);
            acc = fmaf(r.w, cp[3], acc);
        }
        out[((size_t)(b * NC + n)) * CC + c] = acc;
    }
}

// ---------------------------------------------------------------------------
extern "C" void kernel_launch(void* const* d_in, const int* in_sizes, int n_in,
                              void* d_out, int out_size, void* d_ws, size_t ws_size,
                              hipStream_t stream) {
    const float* img  = (const float*)d_in[0];   // [8,2048,14,14]
    const float* word = (const float*)d_in[1];   // [80,300]
    const float* W1   = (const float*)d_in[2];   // [1024,2048]
    const float* W2   = (const float*)d_in[3];   // [1024,300]
    const float* W3   = (const float*)d_in[4];   // [1024,1024]
    const float* b3   = (const float*)d_in[5];   // [1024]
    const float* Wa   = (const float*)d_in[6];   // [1,1024]
    const float* ba   = (const float*)d_in[7];   // [1]
    float* out = (float*)d_out;

    // Workspace carve-up (256B aligned)
    uintptr_t p = (uintptr_t)d_ws;
    auto carve = [&](size_t bytes) { uintptr_t q = p; p += (bytes + 255) & ~(size_t)255; return q; };
    __bf16* img_t = (__bf16*)carve((size_t)SBP * CC * sizeof(__bf16));   // 6.55 MB
    __bf16* W1b   = (__bf16*)carve((size_t)DD * CC * sizeof(__bf16));    // 4.19 MB
    float*  f_wh  = (float*) carve((size_t)SBP * DD * sizeof(float));    // 6.55 MB
    float*  f_wd  = (float*) carve((size_t)NC * DD * sizeof(float));     // 0.33 MB
    float*  v     = (float*) carve((size_t)DD * sizeof(float));
    float*  c0    = (float*) carve(sizeof(float));
    float*  coef  = (float*) carve((size_t)SS * BB * NC * sizeof(float)); // 0.50 MB

    // Stage 0: tiny prep
    k_prep_v  <<<(DD + 255) / 256, 256, 0, stream>>>(W3, Wa, v);
    k_prep_c0 <<<1, 32, 0, stream>>>(b3, Wa, ba, c0);
    k_prep_fwd<<<(NC * DD + 255) / 256, 256, 0, stream>>>(word, W2, f_wd);

    // Stage 1: pack bf16 operands
    k_pack_w1 <<<(DD * CC + 255) / 256, 256, 0, stream>>>(W1, W1b);
    {
        dim3 g((SS + 31) / 32, CC / 32, BB), blk(32, 8);
        k_pack_img<<<g, blk, 0, stream>>>(img, img_t);
    }

    // Stage 2: WMMA GEMM f_wh = img_t @ W1^T
    {
        dim3 g(DD / 128, SBP / 64);
        k_gemm_fwh<<<g, 256, 0, stream>>>(img_t, W1b, f_wh);
    }

    // Stage 3: fused tanh-dot -> coef (dominant: hardware v_tanh_f32)
    k_coef<<<SB, 256, 0, stream>>>(f_wh, f_wd, v, c0, coef);

    // Stage 4: softmax over spatial dim
    k_softmax<<<(BB * NC + 127) / 128, 128, 0, stream>>>(coef);

    // Stage 5: weighted sum -> out[b,n,c]
    {
        dim3 g(NC, BB);
        k_out<<<g, 256, 0, stream>>>(img, coef, out);
    }
    (void)in_sizes; (void)n_in; (void)out_size; (void)ws_size;
}
